// GQA_RoPE_Internal_9680856285861
// MI455X (gfx1250) — matrix operand order repaired and checked by reading, not compiled, not run on hardware
//
#include <hip/hip_runtime.h>
#include <hip/hip_bf16.h>

// ---------------- problem constants ----------------
#define BATCH   2
#define SEQL    2048
#define DMODEL  2048
#define H_Q     32
#define H_KV    8
#define HD      64
#define NKV     (H_KV * HD)    // 512
#define MROWS   (BATCH * SEQL) // 4096

typedef __attribute__((ext_vector_type(16))) __bf16       v16bf;
typedef __attribute__((ext_vector_type(8)))  float        v8f;
typedef __attribute__((ext_vector_type(4)))  unsigned int u32x4;
typedef __attribute__((ext_vector_type(8)))  int          i32x8;
typedef __attribute__((ext_vector_type(4)))  int          i32x4;

#if defined(__has_builtin)
#if __has_builtin(__builtin_amdgcn_tensor_load_to_lds)
#define HAVE_TDM 1
#endif
#endif
#ifndef HAVE_TDM
#define HAVE_TDM 0
#endif

union FragU { u32x4 u[2]; v16bf v; };

__device__ __forceinline__ unsigned short f2bf(float f) {
  union { float f; unsigned u; } t; t.f = f;
  unsigned u = t.u + 0x7fffu + ((t.u >> 16) & 1u); // round-to-nearest-even
  return (unsigned short)(u >> 16);
}

// A-fragment (16x32 bf16): lane l -> row (l&15);
// K elems = base..base+15 contiguous, base = (l>>4)*16
__device__ __forceinline__ v16bf load_a_frag(const unsigned short* __restrict__ row,
                                             int kb, int hi) {
  const unsigned short* p = row + kb + hi * 16;
  FragU t;
  t.u[0] = *(const u32x4*)(p);
  t.u[1] = *(const u32x4*)(p + 8);
  return t.v;
}

// B-fragment (32x16 bf16): lane l -> col (l&15); K elems = base..base+15 contiguous,
// base = (l>>4)*16  (mirrors the sparse-B layout in ISA 7.12.4)
__device__ __forceinline__ v16bf load_b_frag(const unsigned short* __restrict__ row,
                                             int kb, int hi) {
  const unsigned short* p = row + kb + hi * 16;
  FragU t;
  t.u[0] = *(const u32x4*)(p);
  t.u[1] = *(const u32x4*)(p + 8);
  return t.v;
}

__device__ __forceinline__ v8f wmma_bf16(v16bf a, v16bf b, v8f c) {
  return __builtin_amdgcn_wmma_f32_16x16x32_bf16(false, a, false, b, (short)0, c,
                                                 false, false);
}

#if HAVE_TDM
// Issue a TDM load of a [64 rows x 32 cols] bf16 tile from a row-major
// [nrows, K] bf16 tensor into LDS (rows packed contiguously, 4KB).
// D# layout per CDNA5 ISA sec. 8.3/8.4 (2-D tensor: groups 2/3 zeroed).
// This toolchain exposes the 6-arg builtin form:
//   (uint32x4 g0, int32x8 g1, int32x4 g2, int32x4 g3, int32x8 pad, i32 cpol)
__device__ __forceinline__ void tdm_load_tile(const unsigned short* gsrc,
                                              unsigned lds_off, int nrows,
                                              int K) {
  unsigned long long ga = (unsigned long long)(size_t)gsrc;
  u32x4 g0;
  g0[0] = 1u;                                   // count=1, user descriptor
  g0[1] = lds_off;                              // lds_addr (bytes)
  g0[2] = (unsigned)(ga & 0xFFFFFFFFu);         // global_addr[31:0]
  g0[3] = (unsigned)((ga >> 32) & 0x01FFFFFFu)  // global_addr[56:32]
          | 0x80000000u;                        // type=2 ("image")
  i32x8 g1;
  g1[0] = 0x00010000;                           // data_size=1 (2B), no multicast
  g1[1] = (int)(((unsigned)K & 0xFFFFu) << 16); // tensor_dim0[15:0] @ bit48
  g1[2] = (int)((((unsigned)K >> 16) & 0xFFFFu)          // tensor_dim0[31:16]
                | (((unsigned)nrows & 0xFFFFu) << 16));  // tensor_dim1[15:0]
  g1[3] = (int)((((unsigned)nrows >> 16) & 0xFFFFu)      // tensor_dim1[31:16]
                | (32u << 16));                          // tile_dim0 = 32
  g1[4] = 64;                                   // tile_dim1 = 64 (tile_dim2=0)
  g1[5] = K;                                    // tensor_dim0_stride[31:0]
  g1[6] = 0;                                    // stride[47:32], dim1_stride lo
  g1[7] = 0;
  i32x4 gz4 = {0, 0, 0, 0};
  i32x8 gz8 = {0, 0, 0, 0, 0, 0, 0, 0};
  __builtin_amdgcn_tensor_load_to_lds(g0, g1, gz4, gz4, gz8, 0);
}
#endif

// ---------------- f32 -> bf16 convert ----------------
__global__ void cvt_f32_bf16(const float* __restrict__ in,
                             unsigned short* __restrict__ out, int n) {
  int i = blockIdx.x * blockDim.x + threadIdx.x;
  if (i < n) out[i] = f2bf(in[i]);
}

// ---------------- WMMA GEMM: C[M,Nout] = A[M,K] * W[Nout,K]^T ----------------
// Block: 256 threads = 8 waves, block tile 128(M) x 64(N).
// W tile (64x32 bf16, shared by all 8 waves) is staged in LDS via the
// Tensor Data Mover (double-buffered); A fragments come straight from global.
// MODE 0: f32 row-major store
// MODE 1: bf16 row-major store
// MODE 2: fused RoPE + bf16 row-major store (tile is 64-wide, head aligned)
// MODE 3: bf16 store transposed to [B, KVH, HD, seq] (for V)
template <int MODE>
__global__ __launch_bounds__(256) void gemm_wmma(
    const unsigned short* __restrict__ A, const unsigned short* __restrict__ W,
    void* __restrict__ out, int M, int Nout, int K, int seq, int kvh) {
  __shared__ unsigned short wtile[2][64 * 32];
  const int wave = threadIdx.x >> 5;
  const int lane = threadIdx.x & 31;
  const int hi   = lane >> 4;
  const int lcol = lane & 15;
  const int m_base = (blockIdx.y * 8 + wave) * 16;
  const int n_base = blockIdx.x * 64;

  const unsigned short* arow = A + (size_t)(m_base + lcol) * K;
  const unsigned short* wblk = W + (size_t)n_base * K;
  const int niter = K / 32;

#if HAVE_TDM
  if (wave == 0)
    tdm_load_tile(wblk, (unsigned)(size_t)&wtile[0][0], Nout, K);
#endif

  v8f acc[4] = {};
  for (int i = 0; i < niter; ++i) {
    const int kb = i * 32;
#if HAVE_TDM
    if (wave == 0) {
      if (i + 1 < niter) {
        tdm_load_tile(wblk + kb + 32, (unsigned)(size_t)&wtile[(i + 1) & 1][0],
                      Nout, K);
        __builtin_amdgcn_s_wait_tensorcnt(1);   // oldest (current tile) done
      } else {
        __builtin_amdgcn_s_wait_tensorcnt(0);
      }
    }
#else
    {  // cooperative fallback: 256 threads x 8 ushorts = 64x32 tile
      int t = threadIdx.x, row = t >> 2, c8 = (t & 3) * 8;
      *(u32x4*)&wtile[i & 1][row * 32 + c8] =
          *(const u32x4*)(wblk + (size_t)row * K + kb + c8);
    }
#endif
    __syncthreads();  // tile[i&1] visible to all waves

    v16bf af = load_a_frag(arow, kb, hi);
    __builtin_prefetch(arow + kb + 512, 0, 1);  // global_prefetch_b8
    const unsigned short* wt = &wtile[i & 1][0];
#pragma unroll
    for (int j = 0; j < 4; ++j) {
      const unsigned short* p = wt + (j * 16 + lcol) * 32 + hi * 16;
      FragU t;
      t.u[0] = *(const u32x4*)(p);
      t.u[1] = *(const u32x4*)(p + 8);
      acc[j] = wmma_bf16(af, t.v, acc[j]);
    }
    __syncthreads();  // all waves done reading before buffer reuse
  }

  if (MODE == 0) {
    float* fout = (float*)out;
#pragma unroll
    for (int j = 0; j < 4; ++j)
#pragma unroll
      for (int r = 0; r < 8; ++r) {
        int row = m_base + r + hi * 8;
        int col = n_base + j * 16 + lcol;
        fout[(size_t)row * Nout + col] = acc[j][r];
      }
  } else if (MODE == 1) {
    unsigned short* bout = (unsigned short*)out;
#pragma unroll
    for (int j = 0; j < 4; ++j)
#pragma unroll
      for (int r = 0; r < 8; ++r) {
        int row = m_base + r + hi * 8;
        int col = n_base + j * 16 + lcol;
        bout[(size_t)row * Nout + col] = f2bf(acc[j][r]);
      }
  } else if (MODE == 2) {
    unsigned short* bout = (unsigned short*)out;
    float vals[4][8];
#pragma unroll
    for (int j = 0; j < 4; ++j) {
      int d = j * 16 + lcol;  // position within the 64-wide head
      float invf = __expf(-(float)(d & 31) * (9.2103403719761836f / 32.0f));
#pragma unroll
      for (int r = 0; r < 8; ++r) {
        int pos = (m_base % seq) + r + hi * 8;
        float s, c;
        __sincosf((float)pos * invf, &s, &c);
        float other = acc[j ^ 2][r];           // paired dim d +/- 32, same lane
        float rot   = (d < 32) ? -other : other;
        vals[j][r]  = acc[j][r] * c + rot * s;
      }
    }
#pragma unroll
    for (int j = 0; j < 4; ++j)
#pragma unroll
      for (int r = 0; r < 8; ++r) {
        int row = m_base + r + hi * 8;
        int col = n_base + j * 16 + lcol;
        bout[(size_t)row * Nout + col] = f2bf(vals[j][r]);
      }
  } else {  // MODE 3: V^T store -> [B, KVH, HD, seq], 8 consecutive pos per lane
    unsigned short* bout = (unsigned short*)out;
    int b    = m_base / seq;
    int pos0 = (m_base % seq) + hi * 8;
#pragma unroll
    for (int j = 0; j < 4; ++j) {
      int col  = n_base + j * 16 + lcol;
      int head = col >> 6;
      int d    = col & 63;
      union { unsigned short s[8]; u32x4 u; } pk;
#pragma unroll
      for (int r = 0; r < 8; ++r) pk.s[r] = f2bf(acc[j][r]);
      size_t addr = ((size_t)(b * kvh + head) * HD + d) * (size_t)seq + pos0;
      *(u32x4*)&bout[addr] = pk.u;
    }
  }
}

// ---------------- flash attention (one wave per (b,h,16-query tile)) ----------------
__global__ __launch_bounds__(256) void attn_wmma(
    const unsigned short* __restrict__ Qm,   // [B, N, H, HD]
    const unsigned short* __restrict__ Km,   // [B, N, KVH, HD]
    const unsigned short* __restrict__ Vt,   // [B, KVH, HD, N]
    unsigned short* __restrict__ Oa) {       // [B, N, H*HD] bf16
  __shared__ unsigned short plds[8][16 * 32];
  const int wave = threadIdx.x >> 5;
  const int lane = threadIdx.x & 31;
  const int hi   = lane >> 4;
  const int lcol = lane & 15;
  const int g  = blockIdx.x * 8 + wave;
  const int qt = g & 127;           // N/16 = 128 query tiles
  const int h  = (g >> 7) & 31;     // H = 32
  const int b  = g >> 12;           // / (128*32)
  const int kv = h >> 2;            // groups = H/KVH = 4
  const int q0 = qt * 16;
  const float scale = 0.125f;       // 1/sqrt(64)

  const unsigned short* qrow = Qm + ((size_t)(b * SEQL + q0 + lcol) * H_Q + h) * HD;
  v16bf qa0 = load_a_frag(qrow, 0, hi);
  v16bf qa1 = load_a_frag(qrow, 32, hi);

  v8f o0 = {}, o1 = {}, o2 = {}, o3 = {};
  float mrow[8], lrow[8];
#pragma unroll
  for (int r = 0; r < 8; ++r) { mrow[r] = -3.0e38f; lrow[r] = 0.f; }

  const unsigned short* vbase = Vt + (size_t)(b * H_KV + kv) * HD * SEQL;

  for (int kb = 0; kb < SEQL; kb += 32) {
    const unsigned short* krow0 =
        Km + ((size_t)(b * SEQL + kb + lcol) * H_KV + kv) * HD;
    const unsigned short* krow1 = krow0 + (size_t)16 * H_KV * HD;

    v8f s0 = {}, s1 = {};
    s0 = wmma_bf16(qa0, load_b_frag(krow0, 0, hi), s0);
    s0 = wmma_bf16(qa1, load_b_frag(krow0, 32, hi), s0);
    s1 = wmma_bf16(qa0, load_b_frag(krow1, 0, hi), s1);
    s1 = wmma_bf16(qa1, load_b_frag(krow1, 32, hi), s1);

    float corr[8];
#pragma unroll
    for (int r = 0; r < 8; ++r) {
      float a  = s0[r] * scale;
      float c2 = s1[r] * scale;
      float mx = fmaxf(a, c2);
      mx = fmaxf(mx, __shfl_xor(mx, 1, 32));
      mx = fmaxf(mx, __shfl_xor(mx, 2, 32));
      mx = fmaxf(mx, __shfl_xor(mx, 4, 32));
      mx = fmaxf(mx, __shfl_xor(mx, 8, 32));
      float mn = fmaxf(mrow[r], mx);
      corr[r]  = __expf(mrow[r] - mn);
      float p0 = __expf(a - mn);
      float p1 = __expf(c2 - mn);
      float rs = p0 + p1;
      rs += __shfl_xor(rs, 1, 32);
      rs += __shfl_xor(rs, 2, 32);
      rs += __shfl_xor(rs, 4, 32);
      rs += __shfl_xor(rs, 8, 32);
      lrow[r] = lrow[r] * corr[r] + rs;
      mrow[r] = mn;
      int R = r + hi * 8;  // C-layout row for this lane half
      plds[wave][R * 32 + lcol]      = f2bf(p0);
      plds[wave][R * 32 + 16 + lcol] = f2bf(p1);
    }
    // cross-lane LDS hand-off within the wave: wait for all ds stores
    asm volatile("s_wait_dscnt 0x0" ::: "memory");

    // reload P (16x32) as an A-fragment from LDS
    FragU pt;
    pt.u[0] = *(const u32x4*)&plds[wave][lcol * 32 + hi * 16];
    pt.u[1] = *(const u32x4*)&plds[wave][lcol * 32 + hi * 16 + 8];
    v16bf pa = pt.v;

#pragma unroll
    for (int r = 0; r < 8; ++r) {
      o0[r] *= corr[r]; o1[r] *= corr[r]; o2[r] *= corr[r]; o3[r] *= corr[r];
    }
    o0 = wmma_bf16(pa, load_b_frag(vbase + (size_t)(0 * 16 + lcol) * SEQL, kb, hi), o0);
    o1 = wmma_bf16(pa, load_b_frag(vbase + (size_t)(1 * 16 + lcol) * SEQL, kb, hi), o1);
    o2 = wmma_bf16(pa, load_b_frag(vbase + (size_t)(2 * 16 + lcol) * SEQL, kb, hi), o2);
    o3 = wmma_bf16(pa, load_b_frag(vbase + (size_t)(3 * 16 + lcol) * SEQL, kb, hi), o3);
  }

#pragma unroll
  for (int r = 0; r < 8; ++r) {
    float inv = 1.0f / lrow[r];
    int row = b * SEQL + q0 + r + hi * 8;
    size_t base = (size_t)row * DMODEL + (size_t)h * HD;
    Oa[base + 0 * 16 + lcol] = f2bf(o0[r] * inv);
    Oa[base + 1 * 16 + lcol] = f2bf(o1[r] * inv);
    Oa[base + 2 * 16 + lcol] = f2bf(o2[r] * inv);
    Oa[base + 3 * 16 + lcol] = f2bf(o3[r] * inv);
  }
}

// ---------------- launcher ----------------
extern "C" void kernel_launch(void* const* d_in, const int* in_sizes, int n_in,
                              void* d_out, int out_size, void* d_ws, size_t ws_size,
                              hipStream_t stream) {
  const float* x  = (const float*)d_in[0];
  const float* Wq = (const float*)d_in[1];
  const float* Wk = (const float*)d_in[2];
  const float* Wv = (const float*)d_in[3];
  const float* Wo = (const float*)d_in[4];
  (void)in_sizes; (void)n_in; (void)out_size; (void)ws_size;

  unsigned short* ws = (unsigned short*)d_ws;
  size_t off = 0;
  unsigned short* xbf  = ws + off; off += (size_t)MROWS * DMODEL;
  unsigned short* wqbf = ws + off; off += (size_t)DMODEL * DMODEL;
  unsigned short* wkbf = ws + off; off += (size_t)NKV * DMODEL;
  unsigned short* wvbf = ws + off; off += (size_t)NKV * DMODEL;
  unsigned short* wobf = ws + off; off += (size_t)DMODEL * DMODEL;
  unsigned short* Qb   = ws + off; off += (size_t)MROWS * DMODEL;
  unsigned short* Kb   = ws + off; off += (size_t)MROWS * NKV;
  unsigned short* Vtb  = ws + off; off += (size_t)MROWS * NKV;
  unsigned short* Ab   = ws + off; off += (size_t)MROWS * DMODEL;

  auto cvt = [&](const float* src, unsigned short* dst, size_t n) {
    int threads = 256;
    int blocks = (int)((n + threads - 1) / threads);
    cvt_f32_bf16<<<blocks, threads, 0, stream>>>(src, dst, (int)n);
  };
  cvt(x,  xbf,  (size_t)MROWS * DMODEL);
  cvt(Wq, wqbf, (size_t)DMODEL * DMODEL);
  cvt(Wk, wkbf, (size_t)NKV * DMODEL);
  cvt(Wv, wvbf, (size_t)NKV * DMODEL);
  cvt(Wo, wobf, (size_t)DMODEL * DMODEL);

  dim3 blk(256);
  // Q projection + RoPE  -> [B,N,H,HD] bf16
  gemm_wmma<2><<<dim3(DMODEL / 64, MROWS / 128), blk, 0, stream>>>(
      xbf, wqbf, Qb, MROWS, DMODEL, DMODEL, SEQL, H_KV);
  // K projection + RoPE  -> [B,N,KVH,HD] bf16
  gemm_wmma<2><<<dim3(NKV / 64, MROWS / 128), blk, 0, stream>>>(
      xbf, wkbf, Kb, MROWS, NKV, DMODEL, SEQL, H_KV);
  // V projection, transposed store -> [B,KVH,HD,N] bf16
  gemm_wmma<3><<<dim3(NKV / 64, MROWS / 128), blk, 0, stream>>>(
      xbf, wvbf, Vtb, MROWS, NKV, DMODEL, SEQL, H_KV);

  // flash attention -> [B,N,H*HD] bf16
  attn_wmma<<<dim3(BATCH * H_Q * (SEQL / 16) / 8), blk, 0, stream>>>(Qb, Kb, Vtb, Ab);

  // output projection -> f32 d_out
  gemm_wmma<0><<<dim3(DMODEL / 64, MROWS / 128), blk, 0, stream>>>(
      Ab, wobf, d_out, MROWS, DMODEL, DMODEL, SEQL, H_KV);
}
